// Qwen3MoeSparseMoeBlock_4260607558196
// MI455X (gfx1250) — matrix operand order repaired and checked by reading, not compiled, not run on hardware
//
#include <hip/hip_runtime.h>
#include <hip/hip_bf16.h>
#include <math.h>

// ---------------- problem constants ----------------
#define En   16
#define Hdim 1024
#define Idim 768
#define Tn   2048
#define MTILE 32
#define MAX_TILES (Tn / MTILE)   // 64: worst case one expert owns every token

#define GP_ELEMS ((size_t)En * Idim * Hdim)   // elems per weight tensor
#define X_ELEMS  ((size_t)Tn * Hdim)

typedef __attribute__((ext_vector_type(16))) _Float16 v16h;
typedef __attribute__((ext_vector_type(8)))  _Float16 v8h;
typedef __attribute__((ext_vector_type(4)))  _Float16 v4h;
typedef __attribute__((ext_vector_type(8)))  float    v8f;

__device__ __forceinline__ v8f zero_v8f() {
  v8f z = {0.f, 0.f, 0.f, 0.f, 0.f, 0.f, 0.f, 0.f};
  return z;
}

// D = A(16x32 f16) * B(32x16 f16) + C, fp32 accumulate
__device__ __forceinline__ v8f wmma16(v16h a, v16h b, v8f c) {
  return __builtin_amdgcn_wmma_f32_16x16x32_f16(
      false, a, false, b, (short)0, c, false, false);
}

// A-fragment: 8 halves at p, 8 halves at p+8 (two b128 loads). Works for
// global (pre-split activation planes) and LDS (hmid chunk) pointers.
template <typename P>
__device__ __forceinline__ v16h a_frag(P p) {
  v8h s0 = *(const v8h*)p;
  v8h s1 = *(const v8h*)(p + 8);
  return __builtin_shufflevector(s0, s1, 0, 1, 2, 3, 4, 5, 6, 7,
                                         8, 9, 10, 11, 12, 13, 14, 15);
}

// B-fragment: 16 contiguous pre-split halves (32B, aligned).
__device__ __forceinline__ v16h load16h(const _Float16* __restrict__ p) {
  return *(const v16h*)p;
}

// Fallback: B-fragment from a global fp32 row -> f16 hi + f16 lo split.
__device__ __forceinline__ void load_b_f32(const float* __restrict__ p,
                                           v16h& hi, v16h& lo) {
#pragma unroll
  for (int i = 0; i < 4; ++i) {
    float4 f = ((const float4*)p)[i];
    _Float16 h0 = (_Float16)f.x; hi[4*i+0] = h0; lo[4*i+0] = (_Float16)(f.x - (float)h0);
    _Float16 h1 = (_Float16)f.y; hi[4*i+1] = h1; lo[4*i+1] = (_Float16)(f.y - (float)h1);
    _Float16 h2 = (_Float16)f.z; hi[4*i+2] = h2; lo[4*i+2] = (_Float16)(f.z - (float)h2);
    _Float16 h3 = (_Float16)f.w; hi[4*i+3] = h3; lo[4*i+3] = (_Float16)(f.w - (float)h3);
  }
}

// ---------------- kernel 0: reset per-expert counters ----------------
__global__ void moe_zero_counts(int* __restrict__ counts) {
  if (threadIdx.x < En) counts[threadIdx.x] = 0;
}

// ---------------- kernel 0b: split fp32 tensor into f16 hi/lo planes ----------------
__global__ __launch_bounds__(256) void moe_split(const float* __restrict__ w,
                                                 _Float16* __restrict__ hi,
                                                 _Float16* __restrict__ lo,
                                                 int n4) {
  int i = blockIdx.x * 256 + threadIdx.x;   // float4 index
  if (i >= n4) return;
  float4 f = ((const float4*)w)[i];
  v4h h, l;
  h[0] = (_Float16)f.x; l[0] = (_Float16)(f.x - (float)h[0]);
  h[1] = (_Float16)f.y; l[1] = (_Float16)(f.y - (float)h[1]);
  h[2] = (_Float16)f.z; l[2] = (_Float16)(f.z - (float)h[2]);
  h[3] = (_Float16)f.w; l[3] = (_Float16)(f.w - (float)h[3]);
  ((v4h*)hi)[i] = h;
  ((v4h*)lo)[i] = l;
}

// ---------------- kernel 1: router (one wave32 per token) ----------------
__global__ __launch_bounds__(32) void moe_router(
    const float* __restrict__ x, const float* __restrict__ gate_w,
    int* __restrict__ counts, int* __restrict__ lists,
    float* __restrict__ wlist) {
  const int t = blockIdx.x;
  const int lane = threadIdx.x;
  const float4* x4 = (const float4*)(x + (size_t)t * Hdim);
  const float4* g4 = (const float4*)gate_w;

  float acc[En];
#pragma unroll
  for (int e = 0; e < En; ++e) acc[e] = 0.f;

#pragma unroll
  for (int c = 0; c < Hdim / 4 / 32; ++c) {
    float4 xv = x4[c * 32 + lane];
#pragma unroll
    for (int e = 0; e < En; ++e) {
      float4 gv = g4[(size_t)e * (Hdim / 4) + c * 32 + lane];
      acc[e] += xv.x * gv.x + xv.y * gv.y + xv.z * gv.z + xv.w * gv.w;
    }
  }
#pragma unroll
  for (int off = 16; off >= 1; off >>= 1) {
#pragma unroll
    for (int e = 0; e < En; ++e) acc[e] += __shfl_xor(acc[e], off, 32);
  }

  if (lane == 0) {
    int e0 = 0; float l0 = acc[0];
#pragma unroll
    for (int e = 1; e < En; ++e) { if (acc[e] > l0) { l0 = acc[e]; e0 = e; } }
    int e1 = (e0 == 0) ? 1 : 0; float l1 = acc[e1];
#pragma unroll
    for (int e = 0; e < En; ++e) {
      if (e == e0) continue;
      if (acc[e] > l1) { l1 = acc[e]; e1 = e; }
    }
    float w0 = 1.0f / (1.0f + __expf(l1 - l0));   // renormalized top-2 softmax
    float w1 = 1.0f - w0;
    int p0 = atomicAdd(&counts[e0], 1);
    lists[e0 * Tn + p0] = t * 2 + 0;
    wlist[e0 * Tn + p0] = w0;
    int p1 = atomicAdd(&counts[e1], 1);
    lists[e1 * Tn + p1] = t * 2 + 1;
    wlist[e1 * Tn + p1] = w1;
  }
}

// ---------------- kernel 2: grouped expert SwiGLU GEMM ----------------
// grid: (MAX_TILES, En), block 512 = 16 wave32. One block = expert x 32-token tile.
// A (activations) comes from pre-split global f16 planes (gathered rows).
// Weights: PSW ? pre-split f16 planes : inline fp32->f16 split.
#define SHPAD 264   // hmid LDS row stride (halves): 256 + 8, bank-friendly

template <bool PSW>
__global__ __launch_bounds__(512) void moe_expert_gemm(
    const _Float16* __restrict__ xhi, const _Float16* __restrict__ xlo,
    const float* __restrict__ gate_proj, const float* __restrict__ up_proj,
    const float* __restrict__ down_proj,
    const _Float16* __restrict__ whi, const _Float16* __restrict__ wlo,
    const int* __restrict__ counts, const int* __restrict__ lists,
    const float* __restrict__ wlist, float* __restrict__ partial) {
  const int e = blockIdx.y;
  const int tileBase = blockIdx.x * MTILE;
  const int cnt = counts[e];
  if (tileBase >= cnt) return;

  __shared__ __align__(16) _Float16 sH_hi[32][SHPAD];
  __shared__ __align__(16) _Float16 sH_lo[32][SHPAD];
  __shared__ int   sTok[32];
  __shared__ int   sSlot[32];
  __shared__ float sWt[32];
  __shared__ int   sValid[32];

  const int tid  = threadIdx.x;
  const int lane = tid & 31;
  const int wid  = tid >> 5;          // 0..15

  if (tid < 32) {
    int idx = tileBase + tid;
    int valid = (idx < cnt) ? 1 : 0;
    int entry = valid ? lists[e * Tn + idx] : 0;
    sTok[tid]   = entry >> 1;
    sSlot[tid]  = entry & 1;
    sWt[tid]    = valid ? wlist[e * Tn + idx] : 0.0f;
    sValid[tid] = valid;
  }
  __syncthreads();

  const int koff  = (lane >= 16) ? 16 : 0;  // A/B fragment K sub-offset
  const int brow  = lane & 15;              // fragment row/col index
  const int bcol0 = (lane >> 4) * 16;       // B fragment K base per half-wave
  const int mAdd  = (lane >= 16) ? 8 : 0;   // C/D fragment M offset

  // per-lane activation row bases for the two M-tiles (tokens 0-15, 16-31)
  const size_t xoff0 = (size_t)sTok[brow] * Hdim + koff;
  const size_t xoff1 = (size_t)sTok[16 + brow] * Hdim + koff;

  // persistent output accumulators: [mtile][ntile], 32 tokens x 64 cols per wave
  v8f oacc[8];
#pragma unroll
  for (int i = 0; i < 8; ++i) oacc[i] = zero_v8f();

  for (int ic = 0; ic < Idim / 256; ++ic) {       // 3 chunks of 256 I-rows
    // ---- Phase A: this wave computes gate & up for I-rows [iBase, iBase+16)
    const int iBase = ic * 256 + wid * 16;
    v8f gacc0 = zero_v8f(), uacc0 = zero_v8f();
    v8f gacc1 = zero_v8f(), uacc1 = zero_v8f();

    const size_t rowOff = (size_t)e * Idim * Hdim +
                          (size_t)(iBase + brow) * Hdim + bcol0;
    const float*    wgRowF  = gate_proj + rowOff;
    const float*    wuRowF  = up_proj + rowOff;
    const _Float16* wgHiRow = whi + rowOff;
    const _Float16* wgLoRow = wlo + rowOff;
    const _Float16* wuHiRow = whi + GP_ELEMS + rowOff;
    const _Float16* wuLoRow = wlo + GP_ELEMS + rowOff;

#pragma unroll 2
    for (int k0 = 0; k0 < Hdim; k0 += 32) {
      v16h a0h = a_frag(xhi + xoff0 + k0);
      v16h a0l = a_frag(xlo + xoff0 + k0);
      v16h a1h = a_frag(xhi + xoff1 + k0);
      v16h a1l = a_frag(xlo + xoff1 + k0);
      v16h bgh, bgl, buh, bul;
      if constexpr (PSW) {
        bgh = load16h(wgHiRow + k0);
        bgl = load16h(wgLoRow + k0);
        buh = load16h(wuHiRow + k0);
        bul = load16h(wuLoRow + k0);
      } else {
        load_b_f32(wgRowF + k0, bgh, bgl);
        load_b_f32(wuRowF + k0, buh, bul);
      }
      // f16x3 split precision; 4 independent accumulator chains interleaved
      gacc0 = wmma16(a0h, bgh, gacc0);
      uacc0 = wmma16(a0h, buh, uacc0);
      gacc1 = wmma16(a1h, bgh, gacc1);
      uacc1 = wmma16(a1h, buh, uacc1);
      gacc0 = wmma16(a0l, bgh, gacc0);
      uacc0 = wmma16(a0l, buh, uacc0);
      gacc1 = wmma16(a1l, bgh, gacc1);
      uacc1 = wmma16(a1l, buh, uacc1);
      gacc0 = wmma16(a0h, bgl, gacc0);
      uacc0 = wmma16(a0h, bul, uacc0);
      gacc1 = wmma16(a1h, bgl, gacc1);
      uacc1 = wmma16(a1h, bul, uacc1);
    }

    // hmid = silu(g) * u -> LDS chunk (32 tokens x 256 i) as f16 hi/lo
#pragma unroll
    for (int r = 0; r < 8; ++r) {
      float g0 = gacc0[r], u0 = uacc0[r];
      float h0 = (g0 / (1.0f + __expf(-g0))) * u0;
      _Float16 p0 = (_Float16)h0;
      sH_hi[r + mAdd][wid * 16 + brow] = p0;
      sH_lo[r + mAdd][wid * 16 + brow] = (_Float16)(h0 - (float)p0);
      float g1 = gacc1[r], u1 = uacc1[r];
      float h1 = (g1 / (1.0f + __expf(-g1))) * u1;
      _Float16 p1 = (_Float16)h1;
      sH_hi[16 + r + mAdd][wid * 16 + brow] = p1;
      sH_lo[16 + r + mAdd][wid * 16 + brow] = (_Float16)(h1 - (float)p1);
    }
    __syncthreads();

    // ---- Phase B: out(32 tokens x 64 cols per wave) += hmid @ Wd_chunk^T
    const int hBase = wid * 64;
#pragma unroll 2
    for (int kk = 0; kk < 256; kk += 32) {
      v16h a0h = a_frag(&sH_hi[brow][kk + koff]);
      v16h a0l = a_frag(&sH_lo[brow][kk + koff]);
      v16h a1h = a_frag(&sH_hi[16 + brow][kk + koff]);
      v16h a1l = a_frag(&sH_lo[16 + brow][kk + koff]);
      const int iGlob = ic * 256 + kk + bcol0;
#pragma unroll
      for (int t = 0; t < 4; ++t) {
        const size_t dOff = (size_t)e * Hdim * Idim +
                            (size_t)(hBase + t * 16 + brow) * Idim + iGlob;
        v16h bh, bl;
        if constexpr (PSW) {
          bh = load16h(whi + 2 * GP_ELEMS + dOff);
          bl = load16h(wlo + 2 * GP_ELEMS + dOff);
        } else {
          load_b_f32(down_proj + dOff, bh, bl);
        }
        oacc[t]     = wmma16(a0h, bh, oacc[t]);
        oacc[4 + t] = wmma16(a1h, bh, oacc[4 + t]);
        oacc[t]     = wmma16(a0l, bh, oacc[t]);
        oacc[4 + t] = wmma16(a1l, bh, oacc[4 + t]);
        oacc[t]     = wmma16(a0h, bl, oacc[t]);
        oacc[4 + t] = wmma16(a1h, bl, oacc[4 + t]);
      }
    }
    __syncthreads();
  }

  // ---- store: partial[(token,slot)] = route_weight * expert_out (exclusive)
#pragma unroll
  for (int mt = 0; mt < 2; ++mt) {
#pragma unroll
    for (int t = 0; t < 4; ++t) {
#pragma unroll
      for (int r = 0; r < 8; ++r) {
        int m = mt * 16 + r + mAdd;
        if (sValid[m]) {
          int h = wid * 64 + t * 16 + brow;
          size_t off = ((size_t)sTok[m] * 2 + (size_t)sSlot[m]) * Hdim + (size_t)h;
          partial[off] = sWt[m] * oacc[mt * 4 + t][r];
        }
      }
    }
  }
}

// ---------------- kernel 3: combine the two expert partials ----------------
__global__ __launch_bounds__(256) void moe_combine(
    const float* __restrict__ partial, float* __restrict__ out) {
  size_t i = (size_t)blockIdx.x * blockDim.x + threadIdx.x;
  const size_t rowQ = Hdim / 4;
  size_t t = i / rowQ;
  size_t r = i - t * rowQ;
  const float4* p4 = (const float4*)partial;
  float4 a = p4[(2 * t) * rowQ + r];
  float4 b = p4[(2 * t + 1) * rowQ + r];
  float4 o;
  o.x = a.x + b.x; o.y = a.y + b.y; o.z = a.z + b.z; o.w = a.w + b.w;
  ((float4*)out)[i] = o;
}

// ---------------- launch ----------------
extern "C" void kernel_launch(void* const* d_in, const int* in_sizes, int n_in,
                              void* d_out, int out_size, void* d_ws, size_t ws_size,
                              hipStream_t stream) {
  const float* x         = (const float*)d_in[0];  // [1,2048,1024]
  const float* gate_w    = (const float*)d_in[1];  // [16,1024]
  const float* gate_proj = (const float*)d_in[2];  // [16,768,1024]
  const float* up_proj   = (const float*)d_in[3];  // [16,768,1024]
  const float* down_proj = (const float*)d_in[4];  // [16,1024,768]
  float* out = (float*)d_out;                      // [1,2048,1024]

  // workspace layout
  char* ws = (char*)d_ws;
  size_t off = 0;
  int*   counts  = (int*)(ws + off);   off += 256;
  int*   lists   = (int*)(ws + off);   off += (size_t)En * Tn * 4;
  float* wlist   = (float*)(ws + off); off += (size_t)En * Tn * 4;
  float* partial = (float*)(ws + off); off += (size_t)Tn * 2 * Hdim * 4;
  _Float16* xhi  = (_Float16*)(ws + off); off += X_ELEMS * 2;
  _Float16* xlo  = (_Float16*)(ws + off); off += X_ELEMS * 2;
  _Float16* whi  = (_Float16*)(ws + off); off += 3 * GP_ELEMS * 2;
  _Float16* wlo  = (_Float16*)(ws + off); off += 3 * GP_ELEMS * 2;
  const bool psw = (ws_size >= off);   // weight planes fit?

  moe_zero_counts<<<1, 32, 0, stream>>>(counts);
  moe_router<<<Tn, 32, 0, stream>>>(x, gate_w, counts, lists, wlist);

  // always pre-split activations (A side)
  {
    const int n4 = (int)(X_ELEMS / 4);
    moe_split<<<(n4 + 255) / 256, 256, 0, stream>>>(x, xhi, xlo, n4);
  }

  dim3 g(MAX_TILES, En);
  if (psw) {
    const int n4 = (int)(GP_ELEMS / 4);
    const int nb = (n4 + 255) / 256;
    moe_split<<<nb, 256, 0, stream>>>(gate_proj, whi, wlo, n4);
    moe_split<<<nb, 256, 0, stream>>>(up_proj,   whi + GP_ELEMS, wlo + GP_ELEMS, n4);
    moe_split<<<nb, 256, 0, stream>>>(down_proj, whi + 2 * GP_ELEMS, wlo + 2 * GP_ELEMS, n4);
    moe_expert_gemm<true><<<g, 512, 0, stream>>>(xhi, xlo, gate_proj, up_proj, down_proj,
                                                 whi, wlo, counts, lists, wlist, partial);
  } else {
    moe_expert_gemm<false><<<g, 512, 0, stream>>>(xhi, xlo, gate_proj, up_proj, down_proj,
                                                  (const _Float16*)nullptr,
                                                  (const _Float16*)nullptr,
                                                  counts, lists, wlist, partial);
  }
  moe_combine<<<(Tn * Hdim / 4) / 256, 256, 0, stream>>>(partial, out);
}
